// SelfAttention_35553739276993
// MI455X (gfx1250) — matrix operand order repaired and checked by reading, not compiled, not run on hardware
//
#include <hip/hip_runtime.h>

// ---------------------------------------------------------------------------
// Self-attention for MI455X (gfx1250): bf16 WMMA flash attention,
// double-buffered LDS pipelines, spill-free register budgets.
// B=4, S=4096, D_IN=1024, HEAD_DIM=128, fp32 inputs/output.
// ---------------------------------------------------------------------------

constexpr int NB  = 4;     // batch
constexpr int SEQ = 4096;  // sequence length
constexpr int DIN = 1024;  // input dim
constexpr int HD  = 128;   // head dim
constexpr float SM_SCALE = 0.08838834764831845f; // 1/sqrt(128)

typedef __attribute__((ext_vector_type(16))) __bf16 bf16x16;
typedef __attribute__((ext_vector_type(8)))  __bf16 bf16x8;
typedef __attribute__((ext_vector_type(8)))  float  f32x8;
typedef __attribute__((ext_vector_type(4)))  float  f32x4;

union B8U { uint4 u; bf16x8 b; };

// D = A*B + C, bf16 inputs, fp32 accumulate. 16x16x32.
__device__ inline f32x8 wmma_bf16(bf16x16 a, bf16x16 b, f32x8 c) {
  return __builtin_amdgcn_wmma_f32_16x16x32_bf16(
      /*neg_a=*/false, a, /*neg_b=*/false, b,
      /*c_mod=*/(short)0, c, /*reuse_a=*/false, /*reuse_b=*/false);
}

// Load 16 contiguous bf16 (32 bytes, 16B aligned).
__device__ inline bf16x16 ld16bf(const __bf16* p) {
  B8U a, b;
  a.u = *(const uint4*)(p);
  b.u = *(const uint4*)(p + 8);
  bf16x16 r;
#pragma unroll
  for (int j = 0; j < 8; ++j) { r[j] = a.b[j]; r[j + 8] = b.b[j]; }
  return r;
}

// A-fragment (16x32 bf16): e0..7 = K[hl*8 ..], e8..15 = K[16+hl*8 ..].
__device__ inline bf16x16 ld_afrag(const __bf16* row, int hl) {
  B8U a, b;
  a.u = *(const uint4*)(row + hl * 8);
  b.u = *(const uint4*)(row + 16 + hl * 8);
  bf16x16 r;
#pragma unroll
  for (int j = 0; j < 8; ++j) { r[j] = a.b[j]; r[j + 8] = b.b[j]; }
  return r;
}

// ---------------------------------------------------------------------------
// Kernel 0: transpose+convert W_{q,k,v} [1024,128] f32 -> wt [3][128][1024] bf16
// ---------------------------------------------------------------------------
__global__ void wt_prep(const float* __restrict__ Wq,
                        const float* __restrict__ Wk,
                        const float* __restrict__ Wv,
                        __bf16* __restrict__ wt) {
  int idx = blockIdx.x * blockDim.x + threadIdx.x;     // 0 .. 3*128*1024-1
  int mat = idx >> 17;                                  // /131072
  int rem = idx & 131071;
  int h = rem >> 10;
  int k = rem & 1023;
  const float* W = (mat == 0) ? Wq : (mat == 1) ? Wk : Wv;
  wt[idx] = (__bf16)W[k * HD + h];
}

// ---------------------------------------------------------------------------
// Kernel 1: QKV projection. grid (S/128, B, 3), block 512 (16 waves).
// Wave w: rows [blk*128 + (w>>1)*16, +16) x cols [(w&1)*64, +64).
// 4 accumulator tiles/wave keeps VGPR pressure low (no spills); W k-chunks
// staged in LDS double-buffered with register fetch-ahead; B fragments
// software-pipelined (load t+1 before WMMA t).
// ---------------------------------------------------------------------------
__global__ __launch_bounds__(512, 1) __attribute__((amdgpu_waves_per_eu(1)))
void qkv_proj(const float* __restrict__ x,
              const __bf16* __restrict__ wt,
              __bf16* __restrict__ qkv) {
  __shared__ __bf16 sW[2][128][72];   // [buf][out col][k in chunk], 144B rows

  const int tid  = threadIdx.x;
  const int wave = tid >> 5;
  const int lane = tid & 31;
  const int hl = lane >> 4;            // half of wave
  const int n  = lane & 15;            // B/C column, A row
  const int b   = blockIdx.y;
  const int mat = blockIdx.z;
  const int row0 = blockIdx.x * 128 + (wave >> 1) * 16;
  const int col0 = (wave & 1) * 64;

  const float*  xrow = x + ((size_t)(b * SEQ) + row0 + n) * DIN;
  const __bf16* wtm  = wt + (size_t)mat * HD * DIN;
  __bf16* outp = qkv + (size_t)mat * ((size_t)NB * SEQ * HD)
                     + (size_t)(b * SEQ) * HD;

  // Cooperative W chunk fetch: 128 rows x 32 bf16 = 512 x uint4, 1/thread.
  const int wr = tid >> 2;
  const int c8 = (tid & 3) * 8;

  f32x8 acc[4];
#pragma unroll
  for (int t = 0; t < 4; ++t) acc[t] = {};

  // ---- prologue: fetch + publish chunk 0 ----
  uint4 wreg = *(const uint4*)(wtm + (size_t)wr * DIN + c8);
  f32x4 araw[4];
  araw[0] = *(const f32x4*)(xrow + hl * 8);
  araw[1] = *(const f32x4*)(xrow + hl * 8 + 4);
  araw[2] = *(const f32x4*)(xrow + 16 + hl * 8);
  araw[3] = *(const f32x4*)(xrow + 16 + hl * 8 + 4);
  *(uint4*)&sW[0][wr][c8] = wreg;
  __syncthreads();

  int cur = 0;
  for (int kc = 0; kc < DIN; kc += 32) {
    const int nxt = kc + 32;
    uint4 wnxt;
    f32x4 anxt[4];
    if (nxt < DIN) {
      // next chunk's global loads; in flight during the WMMAs below
      wnxt = *(const uint4*)(wtm + (size_t)wr * DIN + nxt + c8);
      anxt[0] = *(const f32x4*)(xrow + nxt + hl * 8);
      anxt[1] = *(const f32x4*)(xrow + nxt + hl * 8 + 4);
      anxt[2] = *(const f32x4*)(xrow + nxt + 16 + hl * 8);
      anxt[3] = *(const f32x4*)(xrow + nxt + 16 + hl * 8 + 4);
    }

    // Convert current A chunk to bf16 fragment.
    bf16x16 a;
#pragma unroll
    for (int j = 0; j < 4; ++j) {
      a[j]      = (__bf16)araw[0][j];
      a[j + 4]  = (__bf16)araw[1][j];
      a[j + 8]  = (__bf16)araw[2][j];
      a[j + 12] = (__bf16)araw[3][j];
    }

    // 4 WMMAs from sW[cur], B fragments software-pipelined.
    bf16x16 bb = ld_afrag(&sW[cur][col0 + n][0], hl);
#pragma unroll
    for (int t = 0; t < 4; ++t) {
      bf16x16 bbn;
      if (t < 3) bbn = ld_afrag(&sW[cur][col0 + (t + 1) * 16 + n][0], hl);
      acc[t] = wmma_bf16(a, bb, acc[t]);
      if (t < 3) bb = bbn;
    }

    if (nxt < DIN) {
      *(uint4*)&sW[cur ^ 1][wr][c8] = wnxt;
#pragma unroll
      for (int i = 0; i < 4; ++i) araw[i] = anxt[i];
    }
    __syncthreads();   // publish next buffer; protect cur from next overwrite
    cur ^= 1;
  }

#pragma unroll
  for (int t = 0; t < 4; ++t) {
#pragma unroll
    for (int r = 0; r < 8; ++r) {
      int row = row0 + hl * 8 + r;                 // C layout: M = half*8 + r
      outp[(size_t)row * HD + col0 + t * 16 + n] = (__bf16)acc[t][r];
    }
  }
}

// ---------------------------------------------------------------------------
// Kernel 2: flash attention. grid (S/128, B), block 256 (8 waves).
// Each wave owns 16 query rows (Q staged in per-wave LDS, reloaded per
// chunk to cut VGPR pressure); 32-key tiles double-buffered in LDS with
// register fetch-ahead and a single barrier per tile.
// ---------------------------------------------------------------------------
__global__ __launch_bounds__(256, 1) __attribute__((amdgpu_waves_per_eu(1)))
void flash_attn(const __bf16* __restrict__ Qb,
                const __bf16* __restrict__ Kb,
                const __bf16* __restrict__ Vb,
                float* __restrict__ out) {
  __shared__ __bf16 sK[2][32][136];   // key tile, row-major      (272B rows)
  __shared__ __bf16 sVT[2][HD][80];   // V tile transposed [h][k] (160B rows)
  __shared__ __bf16 sP[8][16][72];    // per-wave P tile 16x32    (144B rows)
  __shared__ __bf16 sQ[8][16][136];   // per-wave Q tile 16x128   (272B rows)

  const int tid  = threadIdx.x;
  const int wave = tid >> 5;
  const int lane = tid & 31;
  const int hl = lane >> 4;
  const int n  = lane & 15;
  const int b  = blockIdx.y;
  const int q0 = blockIdx.x * 128 + wave * 16;

  const __bf16* kbase = Kb + (size_t)(b * SEQ) * HD;
  const __bf16* vbase = Vb + (size_t)(b * SEQ) * HD;

  // Stage this wave's Q tile (16 x 128) in LDS. Same-wave produce/consume:
  // LDS ops are in-order per wave, no barrier required.
  {
    const __bf16* qg = Qb + ((size_t)(b * SEQ) + q0) * HD;
#pragma unroll
    for (int i = 0; i < 8; ++i) {
      int idx = lane + i * 32;             // 0..255 chunks of 8
      int qr = idx >> 4, qc8 = (idx & 15) * 8;
      uint4 v = *(const uint4*)(qg + (size_t)qr * HD + qc8);
      *(uint4*)&sQ[wave][qr][qc8] = v;
    }
  }

  f32x8 accO[8];
#pragma unroll
  for (int t = 0; t < 8; ++t) accO[t] = {};
  float mrow[8], lrow[8];
#pragma unroll
  for (int r = 0; r < 8; ++r) { mrow[r] = -__builtin_inff(); lrow[r] = 0.0f; }

  // ---- prologue: fetch + store key/value tile 0 ----
  uint4 kreg[2], vreg[2];
#pragma unroll
  for (int i = 0; i < 2; ++i) {
    int idx = tid + i * 256, kr = idx >> 4, c8 = (idx & 15) * 8;
    kreg[i] = *(const uint4*)(kbase + (size_t)kr * HD + c8);
    vreg[i] = *(const uint4*)(vbase + (size_t)kr * HD + c8);
  }
#pragma unroll
  for (int i = 0; i < 2; ++i) {
    int idx = tid + i * 256, kr = idx >> 4, c8 = (idx & 15) * 8;
    *(uint4*)&sK[0][kr][c8] = kreg[i];
    B8U vv; vv.u = vreg[i];
#pragma unroll
    for (int j = 0; j < 8; ++j) sVT[0][c8 + j][kr] = vv.b[j];
  }
  __syncthreads();

  int cur = 0;
  for (int kt = 0; kt < SEQ; kt += 32) {
    const int nxt = kt + 32;
    if (nxt < SEQ) {
      // fetch next tile; loads in flight during the 16 WMMAs + softmax below
#pragma unroll
      for (int i = 0; i < 2; ++i) {
        int idx = tid + i * 256, kr = idx >> 4, c8 = (idx & 15) * 8;
        kreg[i] = *(const uint4*)(kbase + (size_t)(nxt + kr) * HD + c8);
        vreg[i] = *(const uint4*)(vbase + (size_t)(nxt + kr) * HD + c8);
      }
    }

    // Scores: two 16x16 tiles (keys kt+n and kt+16+n), K(h)=128 in 4 chunks.
    f32x8 s0 = {}, s1 = {};
#pragma unroll
    for (int c = 0; c < 4; ++c) {
      bf16x16 qa  = ld16bf(&sQ[wave][n][c * 32 + hl * 16]);
      bf16x16 bk0 = ld16bf(&sK[cur][n][c * 32 + hl * 16]);
      bf16x16 bk1 = ld16bf(&sK[cur][16 + n][c * 32 + hl * 16]);
      s0 = wmma_bf16(qa, bk0, s0);
      s1 = wmma_bf16(qa, bk1, s1);
    }

    // Online softmax. Lane holds rows M = hl*8 + r, column n.
#pragma unroll
    for (int r = 0; r < 8; ++r) {
      float v0 = s0[r] * SM_SCALE;
      float v1 = s1[r] * SM_SCALE;
      float mx = fmaxf(v0, v1);
      mx = fmaxf(mx, __shfl_xor(mx, 1, 32));
      mx = fmaxf(mx, __shfl_xor(mx, 2, 32));
      mx = fmaxf(mx, __shfl_xor(mx, 4, 32));
      mx = fmaxf(mx, __shfl_xor(mx, 8, 32));   // reduce within half (16 cols)
      float nm = fmaxf(mrow[r], mx);
      float alpha = __expf(mrow[r] - nm);
      float p0 = __expf(v0 - nm);
      float p1 = __expf(v1 - nm);
      float ps = p0 + p1;
      ps += __shfl_xor(ps, 1, 32);
      ps += __shfl_xor(ps, 2, 32);
      ps += __shfl_xor(ps, 4, 32);
      ps += __shfl_xor(ps, 8, 32);
      lrow[r] = lrow[r] * alpha + ps;
      mrow[r] = nm;
#pragma unroll
      for (int t = 0; t < 8; ++t) accO[t][r] *= alpha;
      int M = hl * 8 + r;
      sP[wave][M][n]      = (__bf16)p0;
      sP[wave][M][16 + n] = (__bf16)p1;
    }

    // P (16x32) as A-fragment, V as B-fragments: O += P * V.
    bf16x16 pa;
    {
      B8U a, c2;
      a.u  = *(const uint4*)&sP[wave][n][hl * 8];
      c2.u = *(const uint4*)&sP[wave][n][16 + hl * 8];
#pragma unroll
      for (int j = 0; j < 8; ++j) { pa[j] = a.b[j]; pa[j + 8] = c2.b[j]; }
    }
    // B fragments software-pipelined: load t+1 before WMMA t.
    bf16x16 bv = ld_afrag(&sVT[cur][n][0], hl);
#pragma unroll
    for (int t = 0; t < 8; ++t) {
      bf16x16 bvn;
      if (t < 7) bvn = ld_afrag(&sVT[cur][(t + 1) * 16 + n][0], hl);
      accO[t] = wmma_bf16(pa, bv, accO[t]);
      if (t < 7) bv = bvn;
    }

    // Store next tile into the other buffer. Safe: its previous readers
    // finished last iteration (barrier below was passed by everyone).
    if (nxt < SEQ) {
#pragma unroll
      for (int i = 0; i < 2; ++i) {
        int idx = tid + i * 256, kr = idx >> 4, c8 = (idx & 15) * 8;
        *(uint4*)&sK[cur ^ 1][kr][c8] = kreg[i];
        B8U vv; vv.u = vreg[i];
#pragma unroll
        for (int j = 0; j < 8; ++j) sVT[cur ^ 1][c8 + j][kr] = vv.b[j];
      }
    }
    __syncthreads();   // publish tile i+1; protect buf[cur] for next overwrite
    cur ^= 1;
  }

  // Epilogue: divide by running sum, store fp32.
  float* orow = out + ((size_t)(b * SEQ) + q0) * HD;
#pragma unroll
  for (int r = 0; r < 8; ++r) {
    float inv = 1.0f / lrow[r];
    int M = hl * 8 + r;
#pragma unroll
    for (int t = 0; t < 8; ++t)
      orow[(size_t)M * HD + t * 16 + n] = accO[t][r] * inv;
  }
}

// ---------------------------------------------------------------------------
extern "C" void kernel_launch(void* const* d_in, const int* in_sizes, int n_in,
                              void* d_out, int out_size, void* d_ws, size_t ws_size,
                              hipStream_t stream) {
  const float* x  = (const float*)d_in[0];
  const float* Wq = (const float*)d_in[1];
  const float* Wk = (const float*)d_in[2];
  const float* Wv = (const float*)d_in[3];
  float* outp = (float*)d_out;

  // Workspace layout (bf16): Q | K | V | Wt   (~12.75 MB total)
  const size_t qkv_elems = (size_t)NB * SEQ * HD;     // 2,097,152 each
  __bf16* qb = (__bf16*)d_ws;
  __bf16* kb = qb + qkv_elems;
  __bf16* vb = kb + qkv_elems;
  __bf16* wt = vb + qkv_elems;                        // 3*128*1024

  wt_prep<<<(3 * HD * DIN) / 256, 256, 0, stream>>>(Wq, Wk, Wv, wt);
  qkv_proj<<<dim3(SEQ / 128, NB, 3), 512, 0, stream>>>(x, wt, qb);
  flash_attn<<<dim3(SEQ / 128, NB), 256, 0, stream>>>(qb, kb, vb, outp);
}
